// BFEnet_5188320494329
// MI455X (gfx1250) — matrix operand order repaired and checked by reading, not compiled, not run on hardware
//
#include <hip/hip_runtime.h>
#include <hip/hip_bf16.h>

// ---------------------------------------------------------------------------
// BFEnet on gfx1250 (MI455X): all GEMM-shaped work via v_wmma_f32_16x16x32_bf16.
// wave32: each wave owns one 16x16 C tile; A/B use contiguous k (16*half + e), C/D per CDNA5 ISA
// 7.12.2. Dominant GEMMs take a double-buffered fast loop (next k-step's raw
// loads issued before current k-step's cvt+wmma -> partial loadcnt waits).
// Edge tiles use branchless clamp+select loads (no exec-mask churn).
// ---------------------------------------------------------------------------

typedef __attribute__((ext_vector_type(16))) __bf16 v16bf;
typedef __attribute__((ext_vector_type(8)))  float  v8f;

union V16U { v16bf v; __bf16 e[16]; };
union V8U  { v8f   v; float  e[8];  };

#define WAVES_PER_BLOCK 4

__device__ __forceinline__ void packA4(V16U& af, const float4& a0, const float4& a1,
                                       const float4& a2, const float4& a3) {
  af.e[0]  = (__bf16)a0.x; af.e[1]  = (__bf16)a0.y; af.e[2]  = (__bf16)a0.z; af.e[3]  = (__bf16)a0.w;
  af.e[4]  = (__bf16)a1.x; af.e[5]  = (__bf16)a1.y; af.e[6]  = (__bf16)a1.z; af.e[7]  = (__bf16)a1.w;
  af.e[8]  = (__bf16)a2.x; af.e[9]  = (__bf16)a2.y; af.e[10] = (__bf16)a2.z; af.e[11] = (__bf16)a2.w;
  af.e[12] = (__bf16)a3.x; af.e[13] = (__bf16)a3.y; af.e[14] = (__bf16)a3.z; af.e[15] = (__bf16)a3.w;
}

// ---------------------------------------------------------------------------
// Generic batched GEMM: C = act(A @ B + bias)
//   A: [M,K] (lda);  B: [K,N] (ldb) or TRANSB -> B[k,n] = Bsrc[n*ldb + k]
//   act: 0 none, 1 tanh, 2 relu.  One wave per 16x16 C tile; grid.y = batch.
// ---------------------------------------------------------------------------
template <int TRANSB>
__global__ void wmma_gemm(const float* __restrict__ A, const float* __restrict__ Bsrc,
                          const float* __restrict__ bias, float* __restrict__ Cd,
                          int M, int N, int K, int lda, int ldb, int ldc,
                          long long sA, long long sB, long long sC, int act)
{
  int ntiles = (N + 15) >> 4;
  int mtiles = (M + 15) >> 4;
  int wid = (int)blockIdx.x * WAVES_PER_BLOCK + ((int)threadIdx.x >> 5);
  if (wid >= mtiles * ntiles) return;

  const float* Ab = A    + (long long)blockIdx.y * sA;
  const float* Bb = Bsrc + (long long)blockIdx.y * sB;
  float*       Cb = Cd   + (long long)blockIdx.y * sC;

  int mt = wid / ntiles, nt = wid % ntiles;
  int tm = mt << 4, tn = nt << 4;
  int lane = (int)threadIdx.x & 31;
  int half = lane >> 4, l15 = lane & 15;

  int  arow = tm + l15; bool av = arow < M;
  int  bcol = tn + l15; bool bv = bcol < N;
  bool fullMN = (tm + 16 <= M) && (tn + 16 <= N);
  bool vecA   = (lda & 3) == 0;
  bool vecB   = (ldb & 3) == 0;

  V8U acc;
#pragma unroll
  for (int i = 0; i < 8; ++i) acc.e[i] = 0.f;

  // ------------------ pipelined fast loop (dominant GEMMs) ------------------
  if (fullMN && vecA && ((K & 31) == 0) && (!TRANSB || vecB)) {
    const float* apc = Ab + (long long)arow * lda + half * 16;
    const float* bpc;
    if (TRANSB) bpc = Bb + (long long)bcol * ldb + half * 16;
    else        bpc = Bb + (long long)(half * 16) * ldb + bcol;

    // stage 0 raw loads
    float4 a0 = *(const float4*)(apc);
    float4 a1 = *(const float4*)(apc + 4);
    float4 a2 = *(const float4*)(apc + 8);
    float4 a3 = *(const float4*)(apc + 12);
    float4 b0, b1, b2, b3;
    float  bs[16];
    if (TRANSB) {
      b0 = *(const float4*)(bpc);
      b1 = *(const float4*)(bpc + 4);
      b2 = *(const float4*)(bpc + 8);
      b3 = *(const float4*)(bpc + 12);
    } else {
#pragma unroll
      for (int e = 0; e < 16; ++e) bs[e] = bpc[(long long)e * ldb];
    }

    for (int ks = 32;; ks += 32) {
      bool more = ks < K;
      float4 na0, na1, na2, na3, nb0, nb1, nb2, nb3;
      float  nbs[16];
      if (more) {                         // issue next k-step's loads first
        const float* ap2 = apc + ks;
        na0 = *(const float4*)(ap2);
        na1 = *(const float4*)(ap2 + 4);
        na2 = *(const float4*)(ap2 + 8);
        na3 = *(const float4*)(ap2 + 12);
        if (TRANSB) {
          const float* bp2 = bpc + ks;
          nb0 = *(const float4*)(bp2);
          nb1 = *(const float4*)(bp2 + 4);
          nb2 = *(const float4*)(bp2 + 8);
          nb3 = *(const float4*)(bp2 + 12);
        } else {
#pragma unroll
          for (int e = 0; e < 16; ++e) nbs[e] = bpc[(long long)(ks + e) * ldb];
        }
      }
      // convert + multiply current k-step
      V16U af, bfr;
      packA4(af, a0, a1, a2, a3);
      if (TRANSB) packA4(bfr, b0, b1, b2, b3);
      else {
#pragma unroll
        for (int e = 0; e < 16; ++e) bfr.e[e] = (__bf16)bs[e];
      }
      acc.v = __builtin_amdgcn_wmma_f32_16x16x32_bf16(false, af.v, false, bfr.v,
                                                      (short)0, acc.v, false, false);
      if (!more) break;
      a0 = na0; a1 = na1; a2 = na2; a3 = na3;
      if (TRANSB) { b0 = nb0; b1 = nb1; b2 = nb2; b3 = nb3; }
      else {
#pragma unroll
        for (int e = 0; e < 16; ++e) bs[e] = nbs[e];
      }
    }
  } else {
    // ------------------- generic loop (edge tiles / K tails) ----------------
    for (int ks = 0; ks < K; ks += 32) {
      V16U af, bfr;
      if (fullMN && (ks + 32 <= K)) {
        const float* ap = Ab + (long long)arow * lda + ks + half * 16;
#pragma unroll
        for (int e = 0; e < 16; ++e)
          af.e[e] = (__bf16)ap[e];
        if (TRANSB) {
          const float* bp = Bb + (long long)bcol * ldb + ks + half * 16;
#pragma unroll
          for (int e = 0; e < 16; ++e) bfr.e[e] = (__bf16)bp[e];
        } else {
          const float* bp = Bb + (long long)(ks + half * 16) * ldb + bcol;
#pragma unroll
          for (int e = 0; e < 16; ++e) bfr.e[e] = (__bf16)bp[(long long)e * ldb];
        }
      } else {
        // branchless clamp + select
#pragma unroll
        for (int e = 0; e < 16; ++e) {
          int ka = ks + half * 16 + e;
          bool oka = av && (ka < K);
          long long ixa = oka ? ((long long)arow * lda + ka) : 0;
          float va = Ab[ixa];
          af.e[e] = (__bf16)(oka ? va : 0.f);

          int kb = ks + half * 16 + e;
          bool okb = bv && (kb < K);
          long long ixb;
          if (TRANSB) ixb = okb ? ((long long)bcol * ldb + kb) : 0;
          else        ixb = okb ? ((long long)kb * ldb + bcol) : 0;
          float vb = Bb[ixb];
          bfr.e[e] = (__bf16)(okb ? vb : 0.f);
        }
      }
      acc.v = __builtin_amdgcn_wmma_f32_16x16x32_bf16(false, af.v, false, bfr.v,
                                                      (short)0, acc.v, false, false);
    }
  }

#pragma unroll
  for (int r = 0; r < 8; ++r) {
    int row = tm + half * 8 + r;   // C/D: VGPR r -> M=r (lanes0-15), M=8+r (lanes16-31)
    int col = tn + l15;
    if (row < M && col < N) {
      float v = acc.e[r];
      if (bias) v += bias[col];
      if (act == 1) v = tanhf(v);
      else if (act == 2) v = fmaxf(v, 0.f);
      Cb[(long long)row * ldc + col] = v;
    }
  }
}

// ---------------------------------------------------------------------------
// Fused conv(1x5,VALID) + relu + maxpool(1x4) as implicit GEMM:
//   M = 256*62*Wpool (pooled positions), N = Cout, K = Cin*5.
//   M and Cout are exact multiples of 16 for all three convs -> no M/N guards.
//   4 WMMA accumulators (one per pool offset); relu(max(acc)+bias) epilogue.
//   in : [256, Cin, 62, Win]   out: [256, Cout, 62, Wpool]
// ---------------------------------------------------------------------------
__global__ void conv_relu_pool(const float* __restrict__ in, const float* __restrict__ w,
                               const float* __restrict__ bias, float* __restrict__ out,
                               int Cin, int Cout, int Win, int Wpool)
{
  const int H = 62, Bsz = 256;
  const int Mtot = Bsz * H * Wpool;
  const int K = Cin * 5;

  int ntiles = Cout >> 4;
  int mtiles = Mtot >> 4;
  int wid = (int)blockIdx.x * WAVES_PER_BLOCK + ((int)threadIdx.x >> 5);
  if (wid >= mtiles * ntiles) return;

  int mt = wid / ntiles, nt = wid % ntiles;
  int tm = mt << 4, tn = nt << 4;
  int lane = (int)threadIdx.x & 31;
  int half = lane >> 4, l15 = lane & 15;

  int arow = tm + l15;
  int b   = arow / (H * Wpool);
  int rem = arow % (H * Wpool);
  int h   = rem / Wpool;
  int wp  = rem % Wpool;
  long long abase = (long long)b * Cin * H * Win + (long long)h * Win + wp * 4;

  int bcol = tn + l15;

  V8U acc[4];
#pragma unroll
  for (int p = 0; p < 4; ++p)
#pragma unroll
    for (int i = 0; i < 8; ++i) acc[p].e[i] = 0.f;

  for (int ks = 0; ks < K; ks += 32) {
    V16U bfr;
    bool fullK = (ks + 32 <= K);
#pragma unroll
    for (int e = 0; e < 16; ++e) {
      int k = ks + half * 16 + e;
      bool ok = fullK || (k < K);
      int kc = ok ? k : 0;
      int c = kc / 5, t = kc - c * 5;
      float vv = w[((long long)bcol * Cin + c) * 5 + t];
      bfr.e[e] = (__bf16)(ok ? vv : 0.f);
    }
#pragma unroll
    for (int p = 0; p < 4; ++p) {
      V16U af;
#pragma unroll
      for (int e = 0; e < 16; ++e) {
        int k = ks + half * 16 + e;
        bool ok = fullK || (k < K);
        int kc = ok ? k : 0;
        int c = kc / 5, t = kc - c * 5;
        float vv = in[abase + (long long)c * H * Win + p + t];
        af.e[e] = (__bf16)(ok ? vv : 0.f);
      }
      acc[p].v = __builtin_amdgcn_wmma_f32_16x16x32_bf16(false, af.v, false, bfr.v,
                                                         (short)0, acc[p].v, false, false);
    }
  }

#pragma unroll
  for (int r = 0; r < 8; ++r) {
    int row = tm + half * 8 + r;
    int col = tn + l15;
    int b2   = row / (H * Wpool);
    int rem2 = row % (H * Wpool);
    int h2   = rem2 / Wpool;
    int wp2  = rem2 % Wpool;
    float vv = fmaxf(fmaxf(acc[0].e[r], acc[1].e[r]),
                     fmaxf(acc[2].e[r], acc[3].e[r]));
    vv = fmaxf(vv + bias[col], 0.f);
    out[(((long long)b2 * Cout + col) * H + h2) * Wpool + wp2] = vv;
  }
}

// ---------------------------------------------------------------------------
// Per-batch adjacency post-processing, in place on adj[256][62][64] (stride 64):
//   row softmax -> keep top-10 per row (earliest-index tie-break) ->
//   diag := 1 -> deg = clip(rowsum,1)^-0.5 -> adj = D^-1/2 A D^-1/2.
//   One block per batch, 64 threads (thread t owns row t<62), LDS for deg[].
// ---------------------------------------------------------------------------
__global__ void adj_post(float* __restrict__ adj)
{
  __shared__ float degs[62];
  int b = blockIdx.x;
  int t = threadIdx.x;
  float v[62];
  float* row = adj + ((long long)b * 62 + t) * 64;
  float dg = 0.f;

  if (t < 62) {
    float mx = -3.4e38f;
    for (int j = 0; j < 62; ++j) { v[j] = row[j]; mx = fmaxf(mx, v[j]); }
    float s = 0.f;
    for (int j = 0; j < 62; ++j) { float e = __expf(v[j] - mx); v[j] = e; s += e; }
    float inv = 1.f / s;
    for (int j = 0; j < 62; ++j) v[j] *= inv;
    unsigned long long sel = 0ull;
    for (int it = 0; it < 10; ++it) {
      float best = -1.f; int bi = 0;
      for (int j = 0; j < 62; ++j)
        if (!((sel >> j) & 1ull) && v[j] > best) { best = v[j]; bi = j; }
      sel |= (1ull << bi);
    }
    for (int j = 0; j < 62; ++j)
      if (!((sel >> j) & 1ull)) v[j] = 0.f;
    v[t] = 1.0f;                               // self-loop
    float rs = 0.f;
    for (int j = 0; j < 62; ++j) rs += v[j];
    dg = rsqrtf(fmaxf(rs, 1.0f));
    degs[t] = dg;
  }
  __syncthreads();
  if (t < 62) {
    for (int j = 0; j < 62; ++j) row[j] = v[j] * dg * degs[j];
  }
}

// ---------------------------------------------------------------------------
// Host orchestration
// ---------------------------------------------------------------------------
extern "C" void kernel_launch(void* const* d_in, const int* in_sizes, int n_in,
                              void* d_out, int out_size, void* d_ws, size_t ws_size,
                              hipStream_t stream) {
  (void)in_sizes; (void)n_in; (void)out_size; (void)ws_size;

  const float* x       = (const float*)d_in[0];
  const float* conv1_w = (const float*)d_in[1];
  const float* conv1_b = (const float*)d_in[2];
  const float* conv2_w = (const float*)d_in[3];
  const float* conv2_b = (const float*)d_in[4];
  const float* conv3_w = (const float*)d_in[5];
  const float* conv3_b = (const float*)d_in[6];
  const float* g_wbn[3] = {(const float*)d_in[7],  (const float*)d_in[11], (const float*)d_in[15]};
  const float* g_bbn[3] = {(const float*)d_in[8],  (const float*)d_in[12], (const float*)d_in[16]};
  const float* g_wg [3] = {(const float*)d_in[9],  (const float*)d_in[13], (const float*)d_in[17]};
  const float* g_bg [3] = {(const float*)d_in[10], (const float*)d_in[14], (const float*)d_in[18]};
  const float* lin_w   = (const float*)d_in[19];
  const float* lin_b   = (const float*)d_in[20];
  float* outp = (float*)d_out;

  // workspace layout (floats); adj padded to row stride 64 for b128 loads
  float* ws   = (float*)d_ws;
  float* h1   = ws;                  // 256*32*62*65 = 33,013,760
  float* h2   = h1 + 33013760LL;     // 256*64*62*15 = 15,237,120
  float* h3   = h2 + 15237120LL;     // 256*128*62*2 =  4,063,232
  float* xa   = h3 + 4063232LL;      // 256*62*64    =  1,015,808
  float* adjb = xa + 1015808LL;      // 256*62*64    =  1,015,808 (stride 64)
  float* xg   = adjb + 1015808LL;    // 256*62*32    =    507,904
  float* zcat = xg + 507904LL;       // 256*186*32   =  1,523,712

  auto gemm = [&](int transB, const float* A, const float* B, const float* bias, float* C,
                  int M, int N, int K, int lda, int ldb, int ldc,
                  long long sA, long long sB, long long sC,
                  int batches, int act) {
    int tiles = ((M + 15) / 16) * ((N + 15) / 16);
    dim3 grid((unsigned)((tiles + WAVES_PER_BLOCK - 1) / WAVES_PER_BLOCK),
              (unsigned)batches, 1);
    dim3 blk(32 * WAVES_PER_BLOCK);
    if (transB)
      wmma_gemm<1><<<grid, blk, 0, stream>>>(A, B, bias, C, M, N, K, lda, ldb, ldc,
                                             sA, sB, sC, act);
    else
      wmma_gemm<0><<<grid, blk, 0, stream>>>(A, B, bias, C, M, N, K, lda, ldb, ldc,
                                             sA, sB, sC, act);
  };
  auto conv = [&](const float* in, const float* w, const float* bias, float* out,
                  int Cin, int Cout, int Win, int Wpool) {
    int tiles = (256 * 62 * Wpool / 16) * (Cout / 16);
    conv_relu_pool<<<dim3((unsigned)((tiles + WAVES_PER_BLOCK - 1) / WAVES_PER_BLOCK)),
                     dim3(32 * WAVES_PER_BLOCK), 0, stream>>>(
        in, w, bias, out, Cin, Cout, Win, Wpool);
  };
  auto gnn = [&](const float* h, int in_f, int stage) {
    // xa = tanh(X @ Wbn + bbn)   [15872, 64]
    gemm(0, h, g_wbn[stage], g_bbn[stage], xa, 15872, 64, in_f, in_f, 64, 64,
         0, 0, 0, 1, /*tanh*/1);
    // xg = X @ Wg                [15872, 32]
    gemm(0, h, g_wg[stage], nullptr, xg, 15872, 32, in_f, in_f, 32, 32,
         0, 0, 0, 1, 0);
    // adj = xa @ xa^T (batched 256), C stride 64 (padded)
    gemm(1, xa, xa, nullptr, adjb, 62, 62, 64, 64, 64, 64,
         3968, 3968, 3968, 256, 0);
    // softmax + top-10 + self-loops + sym-norm (in place, stride 64)
    adj_post<<<dim3(256), dim3(64), 0, stream>>>(adjb);
    // zcat[:, stage*62:(stage+1)*62, :] = relu(adj @ xg + bg), batched
    gemm(0, adjb, xg, g_bg[stage], zcat + (long long)stage * 62 * 32,
         62, 32, 62, 64, 32, 32, 3968, 1984, 5952, 256, /*relu*/2);
  };

  // stage 1
  conv(x,  conv1_w, conv1_b, h1, 1,  32, 264, 65);
  gnn(h1, 65 * 32, 0);                               // [.,62,2080] view of h1
  // stage 2
  conv(h1, conv2_w, conv2_b, h2, 32, 64, 65, 15);
  gnn(h2, 15 * 64, 1);                               // [.,62,960] view of h2
  // stage 3
  conv(h2, conv3_w, conv3_b, h3, 64, 128, 15, 2);
  gnn(h3, 2 * 128, 2);                               // [.,62,256] view of h3
  // final linear: out = zcat @ lin_w + lin_b   [256, 62]
  gemm(0, zcat, lin_w, lin_b, outp, 256, 62, 5952, 5952, 62, 62, 0, 0, 0, 1, 0);
}
